// DigitCaps_10093173145982
// MI455X (gfx1250) — matrix operand order repaired and checked by reading, not compiled, not run on hardware
//
#include <hip/hip_runtime.h>
#include <math.h>

// CapsNet dynamic routing for MI455X (gfx1250, wave32, WMMA).
//
// pred (536 MB) is never materialized. Each routing round recomputes it with
// v_wmma_f32_16x16x32_f16 and fuses agreement -> logit update -> softmax(N)
// -> weighted accumulation. w stays L2-resident across all 4 passes.
//
// Round-3 changes:
//  * w and x are converted to f16 once per launch (workspace, 71 MB, L2
//    resident). The hot loop loads A fragments with a single
//    global_load_b128 -> no v_cvt chains and no WMMA->VALU WAR v_nops.
//  * Pass 0 packs two p's per WMMA (K=32 fully used: K0-15 = p, K16-31 = p+1,
//    both A and B), halving pass-0 WMMA count and loop trips.

typedef __attribute__((ext_vector_type(16))) _Float16 v16h;
typedef __attribute__((ext_vector_type(8)))  _Float16 v8h;
typedef __attribute__((ext_vector_type(8)))  float    v8f;

namespace caps {
constexpr int B  = 64;
constexpr int P  = 2048;
constexpr int N  = 32;
constexpr int D  = 32;
constexpr int PD = 16;
constexpr int PCHUNK = 16;            // p's per workgroup
constexpr int SROWS  = B * N;         // 2048 (b,n) rows of s/v
constexpr int VPAD   = 36;            // padded d-stride: 16 distinct LDS banks
}

// ---------------------------------------------------------------------------
// One fused routing pass (ITER = 0..3).
//   ITER==0 : c = 1/32 uniform; two p's packed per WMMA; C accumulated.
//   ITER==1 : agreement vs v0, write b1
//   ITER==2 : read b1, +agreement vs v1, write b2
//   ITER==3 : read b2, +agreement vs v2 (no write)
// Wave owns m-tiles tt = wave*8 .. wave*8+7  (n = tt/2, d-half = (tt&1)*16).
// C layout: lanes 0-15 = col b, VGPR j = row d j; lanes 16-31 = row j+8.
// A layout: lanes 0-15 elems 0-7 = K0-7, elems 8-15 = K16-23;
//           lanes 16-31 elems 0-7 = K8-15, elems 8-15 = K24-31.
// B layout: same as A, lanes 0-15 = K0-7,K16-23; lanes 16-31 = K8-15,K24-31 (col = lane%16).
// ---------------------------------------------------------------------------
template<int ITER>
__global__ __launch_bounds__(256)
void routing_pass(const _Float16* __restrict__ xh,   // (B,P,16) f16
                  const _Float16* __restrict__ wh,   // (P,N*D,16) f16
                  const float*    __restrict__ vin,  // (B,N,D)
                  float*          __restrict__ blog, // (B,P,N) routing logits
                  float*          __restrict__ sout) // (B,N,D), f32 atomics
{
    using namespace caps;
    __shared__ __align__(16) float v_sh[N * 16 * VPAD];   // [n][bcol][d(+pad)]
    __shared__ float psum_sh[16 * 8];                     // [bcol][wave]

    const int tid   = threadIdx.x;
    const int wave  = tid >> 5;
    const int lane  = tid & 31;
    const int col   = lane & 15;                   // b column within tile
    const bool hi   = lane >= 16;                  // upper row half
    const int qoff  = hi ? 8 : 0;                  // K sub-offset for A rows
    const int btile = blockIdx.y;
    const int bg    = btile * 16 + col;            // global b for this lane
    const int p0    = blockIdx.x * PCHUNK;
    const int n0    = wave * 4;                    // wave covers n0..n0+3

    if (ITER > 0) {
        // stage v for this b-tile into LDS (coalesced global read)
        for (int i = tid; i < 16 * N * D; i += 256) {
            const int c = i >> 10;                 // bcol
            const int n = (i >> 5) & 31;
            const int d = i & 31;
            v_sh[(n * 16 + c) * VPAD + d] = vin[(size_t)btile * 16384 + i];
        }
        __syncthreads();
    }

    const v8h zh = {};                             // loop-invariant zero half

    v8f sacc[8];
    #pragma unroll
    for (int t = 0; t < 8; ++t) sacc[t] = (v8f){0,0,0,0,0,0,0,0};

    if (ITER == 0) {
        // ---------------- pass 0: two p's per WMMA, full K=32 -------------
        for (int pi = 0; pi < PCHUNK; pi += 2) {
            const int p = p0 + pi;
            // B: elems 0-7 = x[p] dims qoff..+7, elems 8-15 = x[p+1] dims qoff..+7 (matches A).
            const v8h* xp = reinterpret_cast<const v8h*>(
                xh + ((size_t)bg * P + p) * PD + qoff);
            const v16h bf = __builtin_shufflevector(
                xp[0], xp[2], 0,1,2,3,4,5,6,7,8,9,10,11,12,13,14,15);
            #pragma unroll
            for (int t = 0; t < 8; ++t) {
                const int m = (wave * 8 + t) * 16 + col;
                const v8h* w0 = reinterpret_cast<const v8h*>(
                    wh + ((size_t)p * (N * D) + m) * PD + qoff);
                const v8h* w1 = reinterpret_cast<const v8h*>(
                    wh + ((size_t)(p + 1) * (N * D) + m) * PD + qoff);
                const v16h af = __builtin_shufflevector(
                    w0[0], w1[0], 0,1,2,3,4,5,6,7,8,9,10,11,12,13,14,15);
                sacc[t] = __builtin_amdgcn_wmma_f32_16x16x32_f16(
                    false, af, false, bf, (short)0, sacc[t], false, false);
            }
        }
    } else {
        // ---------------- passes 1-3: per-p pred + routing -----------------
        for (int pi = 0; pi < PCHUNK; ++pi) {
            const int p = p0 + pi;

            // B fragment: elems 0-7 = x[p] dims qoff..+7 (matches A); elems 8-15
            // multiplied by zero-padded A half, so zero is fine.
            v16h bf = {};
            {
                const v8h* xp = reinterpret_cast<const v8h*>(
                    xh + ((size_t)bg * P + p) * PD + qoff);
                bf = __builtin_shufflevector(
                    xp[0], zh, 0,1,2,3,4,5,6,7,8,9,10,11,12,13,14,15);
            }

            v8f pred[8];
            #pragma unroll
            for (int t = 0; t < 8; ++t) {
                const int m = (wave * 8 + t) * 16 + col;
                const v8h* wp = reinterpret_cast<const v8h*>(
                    wh + ((size_t)p * (N * D) + m) * PD + qoff);
                const v16h af = __builtin_shufflevector(
                    wp[0], zh, 0,1,2,3,4,5,6,7,8,9,10,11,12,13,14,15);
                v8f c0 = (v8f){0,0,0,0,0,0,0,0};
                pred[t] = __builtin_amdgcn_wmma_f32_16x16x32_f16(
                    false, af, false, bf, (short)0, c0, false, false);
            }

            // agreement <pred, v> (v from LDS, b128 reads; xor-16 joins halves)
            float agr[4];
            #pragma unroll
            for (int nl = 0; nl < 4; ++nl) {
                const int n = n0 + nl;
                float a = 0.0f;
                #pragma unroll
                for (int h = 0; h < 2; ++h) {
                    const int t     = 2 * nl + h;
                    const int dbase = h * 16 + qoff;
                    const float4* vp = reinterpret_cast<const float4*>(
                        &v_sh[(n * 16 + col) * VPAD + dbase]);
                    float4 va = vp[0], vb = vp[1];
                    a += pred[t][0]*va.x + pred[t][1]*va.y
                       + pred[t][2]*va.z + pred[t][3]*va.w
                       + pred[t][4]*vb.x + pred[t][5]*vb.y
                       + pred[t][6]*vb.z + pred[t][7]*vb.w;
                }
                a += __shfl_xor(a, 16, 32);
                agr[nl] = a;
            }

            // routing logits
            const size_t bi = ((size_t)bg * P + p) * N + n0;
            if (ITER >= 2) {
                #pragma unroll
                for (int nl = 0; nl < 4; ++nl) agr[nl] += blog[bi + nl];
            }
            if (ITER <= 2) {
                if (!hi) {
                    #pragma unroll
                    for (int nl = 0; nl < 4; ++nl) blog[bi + nl] = agr[nl];
                }
            }

            // softmax over N=32 (per-wave partial exp-sums through LDS)
            float e[4], ps = 0.0f;
            #pragma unroll
            for (int nl = 0; nl < 4; ++nl) { e[nl] = __expf(agr[nl]); ps += e[nl]; }
            if (!hi) psum_sh[col * 8 + wave] = ps;
            __syncthreads();
            float tot = 0.0f;
            #pragma unroll
            for (int wv = 0; wv < 8; ++wv) tot += psum_sh[col * 8 + wv];
            __syncthreads();
            const float inv = 1.0f / tot;

            // s += c * pred
            #pragma unroll
            for (int t = 0; t < 8; ++t) {
                const float cc = e[t >> 1] * inv;
                #pragma unroll
                for (int j = 0; j < 8; ++j) sacc[t][j] += cc * pred[t][j];
            }
        }
    }

    // flush chunk-partial s (distinct (b,n,d) per lane-half)
    const float fscale = (ITER == 0) ? (1.0f / 32.0f) : 1.0f;
    #pragma unroll
    for (int t = 0; t < 8; ++t) {
        const int n     = n0 + (t >> 1);
        const int dbase = (t & 1) * 16 + qoff;
        float* sp = sout + ((size_t)bg * N + n) * D + dbase;
        #pragma unroll
        for (int j = 0; j < 8; ++j) atomicAdd(sp + j, fscale * sacc[t][j]);
    }
}

// ---------------------------------------------------------------------------
// One-time f32 -> f16 conversion, 8 elements / thread (2x b128 in, 1x b128 out)
__global__ __launch_bounds__(256)
void cvt_f16_kernel(const float* __restrict__ in, _Float16* __restrict__ out,
                    int n8)
{
    const int i = blockIdx.x * 256 + threadIdx.x;
    if (i < n8) {
        const float4* p = reinterpret_cast<const float4*>(in) + (size_t)2 * i;
        float4 a = p[0], b = p[1];
        v8h o;
        o[0]=(_Float16)a.x; o[1]=(_Float16)a.y;
        o[2]=(_Float16)a.z; o[3]=(_Float16)a.w;
        o[4]=(_Float16)b.x; o[5]=(_Float16)b.y;
        o[6]=(_Float16)b.z; o[7]=(_Float16)b.w;
        reinterpret_cast<v8h*>(out)[i] = o;
    }
}

__global__ void zero_kernel(float* __restrict__ p, int n)
{
    int i = blockIdx.x * blockDim.x + threadIdx.x;
    if (i < n) p[i] = 0.0f;
}

// v = squash(s): one wave per (b,n) row, lane = d.
__global__ __launch_bounds__(256)
void squash_kernel(const float* __restrict__ s, float* __restrict__ out)
{
    const int row  = blockIdx.x * 8 + (threadIdx.x >> 5);   // < B*N exact
    const int lane = threadIdx.x & 31;
    const float val = s[(size_t)row * 32 + lane];
    float sq = val * val;
    #pragma unroll
    for (int off = 16; off > 0; off >>= 1) sq += __shfl_xor(sq, off, 32);
    const float scale = (sq / (1.0f + sq)) * rsqrtf(sq + 1e-7f);
    out[(size_t)row * 32 + lane] = val * scale;
}

// ---------------------------------------------------------------------------
extern "C" void kernel_launch(void* const* d_in, const int* in_sizes, int n_in,
                              void* d_out, int out_size, void* d_ws, size_t ws_size,
                              hipStream_t stream)
{
    using namespace caps;
    const float* x = (const float*)d_in[0];      // (64, 2048, 16)
    const float* w = (const float*)d_in[1];      // (2048, 32, 32, 16)
    (void)in_sizes; (void)n_in; (void)out_size; (void)ws_size;

    // workspace layout (~85 MB)
    float*    s_buf = (float*)d_ws;                          //  65536 f32
    float*    v_buf = s_buf + (size_t)B * N * D;             //  65536 f32
    float*    b_buf = v_buf + (size_t)B * N * D;             // 4194304 f32
    _Float16* w_h   = (_Float16*)(b_buf + (size_t)B * P * N);// 33.5M f16
    _Float16* x_h   = w_h + (size_t)P * N * D * PD;          //  2.1M f16

    const int wn8 = (P * N * D * PD) / 8;        // 4194304
    const int xn8 = (B * P * PD) / 8;            //  262144
    cvt_f16_kernel<<<(wn8 + 255) / 256, 256, 0, stream>>>(w, w_h, wn8);
    cvt_f16_kernel<<<(xn8 + 255) / 256, 256, 0, stream>>>(x, x_h, xn8);

    const dim3 grid(P / PCHUNK, B / 16);         // 128 x 4 workgroups
    const int s_elems = B * N * D;

    for (int pass = 0; pass < 4; ++pass) {
        zero_kernel<<<(s_elems + 255) / 256, 256, 0, stream>>>(s_buf, s_elems);
        switch (pass) {
        case 0: routing_pass<0><<<grid, 256, 0, stream>>>(x_h, w_h, v_buf, b_buf, s_buf); break;
        case 1: routing_pass<1><<<grid, 256, 0, stream>>>(x_h, w_h, v_buf, b_buf, s_buf); break;
        case 2: routing_pass<2><<<grid, 256, 0, stream>>>(x_h, w_h, v_buf, b_buf, s_buf); break;
        default: routing_pass<3><<<grid, 256, 0, stream>>>(x_h, w_h, v_buf, b_buf, s_buf); break;
        }
        float* dst = (pass == 3) ? (float*)d_out : v_buf;
        squash_kernel<<<SROWS / 8, 256, 0, stream>>>(s_buf, dst);
    }
}
